// JambaAttentionDecoderLayer_9156870275177
// MI455X (gfx1250) — matrix operand order repaired and checked by reading, not compiled, not run on hardware
//
#include <hip/hip_runtime.h>
#include <hip/hip_bf16.h>
#include <math.h>

#define S_   2048
#define H_   1024
#define NH_  16
#define NKV_ 4
#define HD_  64
#define I_   2816
#define E_   8
#define WIN_ 512
#define QKV_N 1536   // (NH + 2*NKV) * HD

typedef _Float16 v16h __attribute__((ext_vector_type(16)));
typedef _Float16 v8h  __attribute__((ext_vector_type(8)));
typedef float    v8f  __attribute__((ext_vector_type(8)));
typedef unsigned int u32x4 __attribute__((ext_vector_type(4)));
typedef int          i32x8 __attribute__((ext_vector_type(8)));
typedef int          i32x4 __attribute__((ext_vector_type(4)));

// exact parameter type of the async-LDS builtins on this toolchain:
// '__attribute__((__vector_size__(4 * sizeof(int)))) int' pointers in
// __device__ (AS1) / __shared__ (AS3) address spaces.
typedef int vi4 __attribute__((vector_size(16)));
typedef __attribute__((address_space(1))) vi4 gvi4;
typedef __attribute__((address_space(3))) vi4 svi4;

#define BM 128
#define BN 64
#define BK 32

#if __has_builtin(__builtin_amdgcn_global_load_async_to_lds_b128)
#define HAVE_ASYNC_LDS 1
#endif

__device__ __forceinline__ v8f wmma_ff16(v16h a, v16h b, v8f c) {
  // D = A(16x32 f16) * B(32x16 f16) + C(16x16 f32)
  return __builtin_amdgcn_wmma_f32_16x16x32_f16(false, a, false, b, (short)0, c,
                                                false, false);
}

__device__ __forceinline__ v16h pack16(const _Float16* p0, const _Float16* p1) {
  v8h a = *(const v8h*)p0;
  v8h b = *(const v8h*)p1;
  return __builtin_shufflevector(a, b, 0,1,2,3,4,5,6,7,8,9,10,11,12,13,14,15);
}

// --- async global->LDS 16B copy (per-lane), ASYNCcnt tracked -----------------
__device__ __forceinline__ void async_copy_b128(const _Float16* g, _Float16* l) {
#ifdef HAVE_ASYNC_LDS
  __builtin_amdgcn_global_load_async_to_lds_b128((gvi4*)g, (svi4*)l, 0, 0);
#else
  *(v8h*)l = *(const v8h*)g;
#endif
}

__device__ __forceinline__ void wait_async0() {
#ifdef HAVE_ASYNC_LDS
#if __has_builtin(__builtin_amdgcn_s_wait_asynccnt)
  __builtin_amdgcn_s_wait_asynccnt(0);
#else
  asm volatile("s_wait_asynccnt 0x0" ::: "memory");
#endif
#endif
}

__device__ __forceinline__ void wait_tensor0() {
#if __has_builtin(__builtin_amdgcn_s_wait_tensorcnt)
  __builtin_amdgcn_s_wait_tensorcnt(0);
#else
  asm volatile("s_wait_tensorcnt 0x0" ::: "memory");
#endif
}

// --- Tensor Data Mover: DMA a BM x BK f16 tile (row stride lda elements) into
// LDS laid out as [BM][BK+8] f16 (pad 16B per 64B row -> pad_interval=16dw,
// pad_amount=4dw). 2-group descriptor (2D tensor), issued by one wave.
__device__ __forceinline__ void tdm_load_a_tile(const _Float16* gsrc,
                                                unsigned lds_off,
                                                unsigned lda_elems) {
  unsigned long long ga = (unsigned long long)(__UINTPTR_TYPE__)gsrc;
  u32x4 g0;
  g0.x = 1u;                                              // count=1, user desc
  g0.y = lds_off;                                         // LDS byte address
  g0.z = (unsigned)ga;                                    // global_addr[31:0]
  g0.w = (unsigned)((ga >> 32) & 0x01FFFFFFull) | (2u << 30);  // addr hi | type=2
  i32x8 g1;
  g1[0] = (1 << 16)      // data_size = 2 bytes
        | (1 << 20)      // pad_enable
        | (3 << 22)      // pad_interval: 16 DWORDs (one 64B tile row)
        | (3 << 25);     // pad_amount: 4 DWORDs (16B)
  g1[1] = (int)(32u << 16);          // tensor_dim0 = 32 (bits 79:48, low part)
  g1[2] = (int)(0u);                 // tensor_dim0 hi, tensor_dim1 lo16 = 0
  g1[3] = (int)(16u | (32u << 16));  // tensor_dim1 = 1<<20 rows; tile_dim0 = 32
  g1[4] = (int)BM;                   // tile_dim1 = 128 rows (tile_dim2 = 0)
  g1[5] = (int)lda_elems;            // tensor_dim0_stride (data_size units)
  g1[6] = 0;
  g1[7] = 0;
  asm volatile("tensor_load_to_lds %0, %1" :: "s"(g0), "s"(g1) : "memory");
}

// ---------------------------------------------------------------------------
// Shared WMMA GEMM core:  C[BMxBN] += A_f16[BMxK] * W_f32[BNxK]^T
// A rows contiguous with lda == K. W converted f32->f16 while staging to LDS.
// 256 threads = 8 waves arranged 4(M) x 2(N); each wave owns a 32x32 subtile.
// USE_TDM: A tile moved by the Tensor Data Mover (dense A only);
// otherwise per-lane async-to-LDS copies (works for gathered rows).
// ---------------------------------------------------------------------------
template <bool USE_TDM>
__device__ __forceinline__ void gemm_core(const _Float16* __restrict__ aRowPtr,
                                          const _Float16* __restrict__ aTileBase,
                                          int lda,
                                          const float* __restrict__ wRowPtr,
                                          int K,
                                          _Float16 (&As)[BM][BK + 8],
                                          _Float16 (&Bs)[BN][BK + 8],
                                          v8f (&acc)[2][2]) {
  const int tid  = threadIdx.x;
  const int lane = tid & 31;
  const int wave = tid >> 5;
  const int wm   = wave >> 1;      // 0..3 along M
  const int wn   = wave & 1;       // 0..1 along N
  const int mrow = lane & 15;
  const int hi   = lane >> 4;

  const int ar = tid >> 1;         // 0..127 : A row in block
  const int ah = tid & 1;          // half of the 32-wide K slice
  const int wr = tid >> 2;         // 0..63  : W row in block
  const int wq = tid & 3;          // quarter (8 floats)

  for (int k0 = 0; k0 < K; k0 += BK) {
    // ---- stage A (f16 copy: TDM or per-lane async) ----
    if (USE_TDM) {
      if (wave == 0) {
        unsigned asOff = (unsigned)(__UINTPTR_TYPE__)
            (__attribute__((address_space(3))) _Float16*)&As[0][0];
        tdm_load_a_tile(aTileBase + k0, asOff, (unsigned)lda);
        wait_tensor0();
      }
    } else {
      async_copy_b128(aRowPtr + k0,     &As[ar][ah * 16]);
      async_copy_b128(aRowPtr + k0 + 8, &As[ar][ah * 16 + 8]);
      wait_async0();
    }
    // ---- stage W (f32 -> f16 convert on the VALU) ----
    {
      const float* p = wRowPtr + k0;             // includes + wq*8
      float4 f0 = *(const float4*)p;
      float4 f1 = *(const float4*)(p + 4);
      v8h wb = { (_Float16)f0.x, (_Float16)f0.y, (_Float16)f0.z, (_Float16)f0.w,
                 (_Float16)f1.x, (_Float16)f1.y, (_Float16)f1.z, (_Float16)f1.w };
      *(v8h*)&Bs[wr][wq * 8] = wb;
      if (k0 + BK < K) __builtin_prefetch(p + BK, 0, 0);  // global_prefetch_b8
    }
    __syncthreads();

    // ---- fragments + WMMA ----
    v16h Af[2], Bf[2];
#pragma unroll
    for (int mt = 0; mt < 2; ++mt) {
      const _Float16* p = &As[wm * 32 + mt * 16 + mrow][0];
      Af[mt] = pack16(p + hi * 8, p + 16 + hi * 8);     // A layout chunks
    }
#pragma unroll
    for (int nt = 0; nt < 2; ++nt) {
      const _Float16* p = &Bs[wn * 32 + nt * 16 + mrow][0];
      Bf[nt] = pack16(p + hi * 8, p + 16 + hi * 8);     // B layout chunks
    }
#pragma unroll
    for (int mt = 0; mt < 2; ++mt)
#pragma unroll
      for (int nt = 0; nt < 2; ++nt)
        acc[mt][nt] = wmma_ff16(Af[mt], Bf[nt], acc[mt][nt]);
    __syncthreads();
  }
}

// ---------------------------------------------------------------------------
__global__ __launch_bounds__(256) void zero_init_kernel(float* __restrict__ out,
                                                        int* __restrict__ counts) {
  size_t i = (size_t)blockIdx.x * 256 + threadIdx.x;
  if (i < (size_t)S_ * H_) out[i] = 0.0f;
  if (blockIdx.x == 0 && threadIdx.x < E_) counts[threadIdx.x] = 0;
}

__global__ __launch_bounds__(256) void rmsnorm_cast_kernel(
    const float* __restrict__ x, const float* __restrict__ w,
    _Float16* __restrict__ out) {
  const int row = blockIdx.x;
  const int t   = threadIdx.x;
  const float* xr = x + (size_t)row * H_;
  float v[4];
  float s = 0.f;
#pragma unroll
  for (int i = 0; i < 4; ++i) { v[i] = xr[t + i * 256]; s += v[i] * v[i]; }
  __shared__ float red[256];
  red[t] = s;
  __syncthreads();
  for (int st = 128; st > 0; st >>= 1) {
    if (t < st) red[t] += red[t + st];
    __syncthreads();
  }
  const float inv = rsqrtf(red[0] * (1.0f / H_) + 1e-5f);
#pragma unroll
  for (int i = 0; i < 4; ++i) {
    int c = t + i * 256;
    out[(size_t)row * H_ + c] = (_Float16)(v[i] * inv * w[c]);
  }
}

// C_f16[M x N] = A_f16 * W^T   (dense A -> TDM staging)
__global__ __launch_bounds__(256) void gemm_f16_kernel(
    const _Float16* __restrict__ A, const float* __restrict__ W,
    _Float16* __restrict__ C, int K, int ldc) {
  __shared__ __align__(16) _Float16 As[BM][BK + 8];
  __shared__ __align__(16) _Float16 Bs[BN][BK + 8];
  const int m0 = blockIdx.x * BM, n0 = blockIdx.y * BN;
  const int tid = threadIdx.x;
  const _Float16* aTileBase = A + (size_t)m0 * K;
  const _Float16* aRowPtr = A + (size_t)(m0 + (tid >> 1)) * K + (tid & 1) * 16;
  const float*    wRowPtr = W + (size_t)(n0 + (tid >> 2)) * K + (tid & 3) * 8;
  v8f acc[2][2] = {};
  gemm_core<true>(aRowPtr, aTileBase, K, wRowPtr, K, As, Bs, acc);
  const int lane = tid & 31, wave = tid >> 5;
  const int wm = wave >> 1, wn = wave & 1, mrow = lane & 15, hi = lane >> 4;
#pragma unroll
  for (int mt = 0; mt < 2; ++mt)
#pragma unroll
    for (int nt = 0; nt < 2; ++nt)
#pragma unroll
      for (int j = 0; j < 8; ++j) {
        int row = m0 + wm * 32 + mt * 16 + j + hi * 8;
        int col = n0 + wn * 32 + nt * 16 + mrow;
        C[(size_t)row * ldc + col] = (_Float16)acc[mt][nt][j];
      }
}

// x2 = A*W^T + residual  (f32 out; dense A -> TDM staging)
__global__ __launch_bounds__(256) void gemm_oproj_kernel(
    const _Float16* __restrict__ A, const float* __restrict__ W,
    const float* __restrict__ residual, float* __restrict__ C, int K) {
  __shared__ __align__(16) _Float16 As[BM][BK + 8];
  __shared__ __align__(16) _Float16 Bs[BN][BK + 8];
  const int m0 = blockIdx.x * BM, n0 = blockIdx.y * BN;
  const int tid = threadIdx.x;
  const _Float16* aTileBase = A + (size_t)m0 * K;
  const _Float16* aRowPtr = A + (size_t)(m0 + (tid >> 1)) * K + (tid & 1) * 16;
  const float*    wRowPtr = W + (size_t)(n0 + (tid >> 2)) * K + (tid & 3) * 8;
  v8f acc[2][2] = {};
  gemm_core<true>(aRowPtr, aTileBase, K, wRowPtr, K, As, Bs, acc);
  const int lane = tid & 31, wave = tid >> 5;
  const int wm = wave >> 1, wn = wave & 1, mrow = lane & 15, hi = lane >> 4;
#pragma unroll
  for (int mt = 0; mt < 2; ++mt)
#pragma unroll
    for (int nt = 0; nt < 2; ++nt)
#pragma unroll
      for (int j = 0; j < 8; ++j) {
        int row = m0 + wm * 32 + mt * 16 + j + hi * 8;
        int col = n0 + wn * 32 + nt * 16 + mrow;
        size_t idx = (size_t)row * H_ + col;
        C[idx] = acc[mt][nt][j] + residual[idx];
      }
}

// ---------------------------------------------------------------------------
// Flash attention, sliding window, one wave per (16-query tile, head).
// qkv f16 rows of 1536: [q 0..1023 | k 1024..1279 | v 1280..1535]
// ---------------------------------------------------------------------------
__global__ __launch_bounds__(32) void attn_kernel(const _Float16* __restrict__ qkv,
                                                  _Float16* __restrict__ o) {
  const int qt = blockIdx.x, h = blockIdx.y;
  const int q0 = qt * 16;
  const int lane = threadIdx.x;
  const int mrow = lane & 15, hi = lane >> 4;
  const int kvh  = h / (NH_ / NKV_);

  __shared__ __align__(16) _Float16 Plds[16][40];

  // Q fragments (A layout), held in regs all kernel
  const _Float16* qp = qkv + (size_t)(q0 + mrow) * QKV_N + h * HD_;
  const v16h Aq0 = pack16(qp + hi * 8,      qp + 16 + hi * 8);
  const v16h Aq1 = pack16(qp + 32 + hi * 8, qp + 48 + hi * 8);

  const _Float16* kbase = qkv + NH_ * HD_ + kvh * HD_;
  const _Float16* vbase = qkv + NH_ * HD_ + NKV_ * HD_ + kvh * HD_;

  const float NEG = -3.0e38f;
  float m_i[8], l_i[8];
  v8f acc[4];
#pragma unroll
  for (int j = 0; j < 8; ++j) { m_i[j] = NEG; l_i[j] = 0.f; }
#pragma unroll
  for (int t = 0; t < 4; ++t) acc[t] = (v8f)0.f;

  int jlo = q0 - (WIN_ - 1);
  if (jlo < 0) jlo = 0;
  jlo &= ~31;
  const int jhi = q0 + 15;

  for (int j0 = jlo; j0 <= jhi; j0 += 32) {
    // ---- scores: Q(16x64) * K^T(64x32) -> two 16x16 f32 tiles ----
    v8f sc[2];
#pragma unroll
    for (int nt = 0; nt < 2; ++nt) {
      int key = j0 + nt * 16 + mrow;
      int keyc = key < S_ - 1 ? key : S_ - 1;
      const _Float16* kp = kbase + (size_t)keyc * QKV_N;
      v16h Bk0 = pack16(kp + hi * 8,      kp + 16 + hi * 8);       // hd 0..31
      v16h Bk1 = pack16(kp + 32 + hi * 8, kp + 48 + hi * 8);       // hd 32..63
      v8f c = (v8f)0.f;
      c = wmma_ff16(Aq1, Bk1, c);
      c = wmma_ff16(Aq0, Bk0, c);
      sc[nt] = c;
    }
    // ---- masked chunk row-max (C layout: lane holds 8 rows of one column) ----
    float cm[8];
#pragma unroll
    for (int j = 0; j < 8; ++j) {
      int row = q0 + j + hi * 8;
      float mx = NEG;
#pragma unroll
      for (int nt = 0; nt < 2; ++nt) {
        int key = j0 + nt * 16 + mrow;
        bool ok = (key <= row) && (row - key < WIN_) && (key < S_);
        float s = ok ? sc[nt][j] * 0.125f : NEG;
        mx = fmaxf(mx, s);
      }
      cm[j] = mx;
    }
#pragma unroll
    for (int off = 1; off < 16; off <<= 1)
#pragma unroll
      for (int j = 0; j < 8; ++j) cm[j] = fmaxf(cm[j], __shfl_xor(cm[j], off));

    // ---- online softmax update (all stats stay in C-layout registers) ----
    float rloc[8];
#pragma unroll
    for (int j = 0; j < 8; ++j) {
      int row = q0 + j + hi * 8;
      float mnew = fmaxf(m_i[j], cm[j]);
      float sca  = __expf(m_i[j] - mnew);   // m_i <= mnew, no overflow
      m_i[j] = mnew;
      l_i[j] *= sca;
#pragma unroll
      for (int t = 0; t < 4; ++t) acc[t][j] *= sca;
      float rs = 0.f;
#pragma unroll
      for (int nt = 0; nt < 2; ++nt) {
        int key = j0 + nt * 16 + mrow;
        bool ok = (key <= row) && (row - key < WIN_) && (key < S_);
        float p = ok ? __expf(sc[nt][j] * 0.125f - mnew) : 0.f;
        rs += p;
        Plds[j + hi * 8][nt * 16 + mrow] = (_Float16)p;
      }
      rloc[j] = rs;
    }
#pragma unroll
    for (int off = 1; off < 16; off <<= 1)
#pragma unroll
      for (int j = 0; j < 8; ++j) rloc[j] += __shfl_xor(rloc[j], off);
#pragma unroll
    for (int j = 0; j < 8; ++j) l_i[j] += rloc[j];
    __syncthreads();

    // ---- P (16x32) A-fragment from LDS ----
    const _Float16* pp = &Plds[mrow][0];
    v16h Ap = pack16(pp + hi * 16, pp + hi * 16 + 8);

    // ---- acc += P * V(32x64): 4 n-tiles of 16 hd columns ----
#pragma unroll
    for (int nt = 0; nt < 4; ++nt) {
      int hd = nt * 16 + mrow;
      v16h Bv;
#pragma unroll
      for (int kk = 0; kk < 16; ++kk) {
        int key = j0 + hi * 16 + kk;
        if (key > S_ - 1) key = S_ - 1;
        Bv[kk] = vbase[(size_t)key * QKV_N + hd];
      }
      acc[nt] = wmma_ff16(Ap, Bv, acc[nt]);
    }
    __syncthreads();
  }

  // ---- epilogue: normalize and store f16 ----
#pragma unroll
  for (int j = 0; j < 8; ++j) {
    float inv = 1.0f / l_i[j];
    int row = q0 + j + hi * 8;
#pragma unroll
    for (int nt = 0; nt < 4; ++nt) {
      int col = h * HD_ + nt * 16 + mrow;
      o[(size_t)row * (NH_ * HD_) + col] = (_Float16)(acc[nt][j] * inv);
    }
  }
}

// ---------------------------------------------------------------------------
// Router: softmax(logits) -> top-2 -> per-expert token lists (deterministic).
// ---------------------------------------------------------------------------
__global__ __launch_bounds__(256) void router_kernel(
    const _Float16* __restrict__ th, const float* __restrict__ rw,
    int* __restrict__ counts, int* __restrict__ list, float* __restrict__ gates) {
  int t = blockIdx.x * 256 + threadIdx.x;
  if (t >= S_) return;
  const _Float16* xr = th + (size_t)t * H_;
  float a[E_] = {};
  for (int i = 0; i < H_; ++i) {
    float x = (float)xr[i];
#pragma unroll
    for (int e = 0; e < E_; ++e) a[e] += x * rw[e * H_ + i];
  }
  float mx = a[0];
#pragma unroll
  for (int e = 1; e < E_; ++e) mx = fmaxf(mx, a[e]);
  float se = 0.f;
  float p[E_];
#pragma unroll
  for (int e = 0; e < E_; ++e) { p[e] = __expf(a[e] - mx); se += p[e]; }
  float invse = 1.0f / se;
#pragma unroll
  for (int e = 0; e < E_; ++e) p[e] *= invse;
  int i1 = 0;
#pragma unroll
  for (int e = 1; e < E_; ++e) if (p[e] > p[i1]) i1 = e;
  int i2 = (i1 == 0) ? 1 : 0;
#pragma unroll
  for (int e = 0; e < E_; ++e) if (e != i1 && p[e] > p[i2]) i2 = e;
  int picks[2] = {i1, i2};
#pragma unroll
  for (int k = 0; k < 2; ++k) {
    int e = picks[k];
    int slot = atomicAdd(&counts[e], 1);
    list[e * S_ + slot]  = t;
    gates[e * S_ + slot] = p[e];
  }
}

__global__ void scan_kernel(const int* __restrict__ counts, int* __restrict__ offsets) {
  if (threadIdx.x == 0 && blockIdx.x == 0) {
    int o = 0;
    for (int e = 0; e < E_; ++e) { offsets[e] = o; o += counts[e]; }
  }
}

// a = silu(g) * u for gathered tokens of one expert (async-LDS A staging).
__global__ __launch_bounds__(256) void gemm_gated_kernel(
    const _Float16* __restrict__ th, const float* __restrict__ ws,
    const int* __restrict__ counts, const int* __restrict__ offsets,
    const int* __restrict__ list, _Float16* __restrict__ abuf) {
  const int e   = blockIdx.z;
  const int cnt = counts[e];
  const int m0  = blockIdx.x * BM;
  if (m0 >= cnt) return;
  const int n0  = blockIdx.y * BN;
  const int tid = threadIdx.x;

  __shared__ __align__(16) _Float16 As[BM][BK + 8];
  __shared__ __align__(16) _Float16 Bs[BN][BK + 8];

  int r   = m0 + (tid >> 1);
  int tok = list[e * S_ + ((r < cnt) ? r : 0)];
  const _Float16* aRowPtr = th + (size_t)tok * H_ + (tid & 1) * 16;
  const float* We = ws + (size_t)e * 2 * I_ * H_;
  const float* wg = We + (size_t)(n0 + (tid >> 2)) * H_ + (tid & 3) * 8;
  const float* wu = We + (size_t)(I_ + n0 + (tid >> 2)) * H_ + (tid & 3) * 8;

  v8f accg[2][2] = {}, accu[2][2] = {};
  gemm_core<false>(aRowPtr, nullptr, 0, wg, H_, As, Bs, accg);
  gemm_core<false>(aRowPtr, nullptr, 0, wu, H_, As, Bs, accu);

  const int lane = tid & 31, wave = tid >> 5;
  const int wm = wave >> 1, wn = wave & 1, mrow = lane & 15, hi = lane >> 4;
  const int rowbase = offsets[e];
#pragma unroll
  for (int mt = 0; mt < 2; ++mt)
#pragma unroll
    for (int nt = 0; nt < 2; ++nt)
#pragma unroll
      for (int j = 0; j < 8; ++j) {
        int slot = m0 + wm * 32 + mt * 16 + j + hi * 8;
        if (slot < cnt) {
          int col = n0 + wn * 32 + nt * 16 + mrow;
          float g = accg[mt][nt][j], u = accu[mt][nt][j];
          float v = g / (1.0f + __expf(-g)) * u;
          abuf[(size_t)(rowbase + slot) * I_ + col] = (_Float16)v;
        }
      }
}

// y = a * w2^T, scaled by gate, atomically accumulated into out (2 adds/elem,
// float add is commutative -> bitwise deterministic).
__global__ __launch_bounds__(256) void gemm_y_kernel(
    const _Float16* __restrict__ abuf, const float* __restrict__ w2s,
    const int* __restrict__ counts, const int* __restrict__ offsets,
    const int* __restrict__ list, const float* __restrict__ gates,
    float* __restrict__ out) {
  const int e   = blockIdx.z;
  const int cnt = counts[e];
  const int m0  = blockIdx.x * BM;
  if (m0 >= cnt) return;
  const int n0  = blockIdx.y * BN;
  const int tid = threadIdx.x;

  __shared__ __align__(16) _Float16 As[BM][BK + 8];
  __shared__ __align__(16) _Float16 Bs[BN][BK + 8];

  int r  = m0 + (tid >> 1);
  int ar = offsets[e] + ((r < cnt) ? r : (cnt - 1));
  const _Float16* aRowPtr = abuf + (size_t)ar * I_ + (tid & 1) * 16;
  const float* wRowPtr = w2s + (size_t)e * H_ * I_ +
                         (size_t)(n0 + (tid >> 2)) * I_ + (tid & 3) * 8;

  v8f acc[2][2] = {};
  gemm_core<false>(aRowPtr, nullptr, 0, wRowPtr, I_, As, Bs, acc);

  const int lane = tid & 31, wave = tid >> 5;
  const int wm = wave >> 1, wn = wave & 1, mrow = lane & 15, hi = lane >> 4;
#pragma unroll
  for (int mt = 0; mt < 2; ++mt)
#pragma unroll
    for (int nt = 0; nt < 2; ++nt)
#pragma unroll
      for (int j = 0; j < 8; ++j) {
        int slot = m0 + wm * 32 + mt * 16 + j + hi * 8;
        if (slot < cnt) {
          int tok = list[e * S_ + slot];
          float g = gates[e * S_ + slot];
          int col = n0 + wn * 32 + nt * 16 + mrow;
          atomicAdd(&out[(size_t)tok * H_ + col], g * acc[mt][nt][j]);
        }
      }
}

// ---------------------------------------------------------------------------
extern "C" void kernel_launch(void* const* d_in, const int* in_sizes, int n_in,
                              void* d_out, int out_size, void* d_ws, size_t ws_size,
                              hipStream_t stream) {
  (void)in_sizes; (void)n_in; (void)out_size; (void)ws_size;
  const float* hidden   = (const float*)d_in[0];
  // d_in[1] = positions (unused; mask is recomputed analytically)
  const float* w_qkv    = (const float*)d_in[2];
  const float* w_o      = (const float*)d_in[3];
  const float* router_w = (const float*)d_in[4];
  const float* ws       = (const float*)d_in[5];
  const float* w2s      = (const float*)d_in[6];
  const float* ln1_w    = (const float*)d_in[7];
  const float* ln2_w    = (const float*)d_in[8];

  float* out  = (float*)d_out;            // MoE output
  float* res2 = out + (size_t)S_ * H_;    // residual2 output

  char* wsb = (char*)d_ws;
  _Float16* xh    = (_Float16*)(wsb);                      //  4 MB
  _Float16* qkvh  = (_Float16*)(wsb + (4u  << 20));        //  6 MB
  _Float16* oh    = (_Float16*)(wsb + (10u << 20));        //  4 MB
  _Float16* th    = (_Float16*)(wsb + (14u << 20));        //  4 MB
  int*      counts  = (int*)(wsb + (18u << 20));
  int*      offsets = counts + 16;
  int*      list    = (int*)(wsb + (18u << 20) + 4096);    // 64 KB
  float*    gates   = (float*)(wsb + (18u << 20) + 4096 + (size_t)E_ * S_ * 4);
  _Float16* abuf    = (_Float16*)(wsb + (19u << 20));      // 4096 x I f16 = 23 MB

  // 1. zero out + router counts
  zero_init_kernel<<<dim3(((size_t)S_ * H_ + 255) / 256), 256, 0, stream>>>(out, counts);
  // 2. rmsnorm1 -> f16
  rmsnorm_cast_kernel<<<dim3(S_), 256, 0, stream>>>(hidden, ln1_w, xh);
  // 3. qkv = x @ w_qkv^T  (f16 out, TDM A staging)
  gemm_f16_kernel<<<dim3(S_ / BM, QKV_N / BN), 256, 0, stream>>>(
      xh, w_qkv, qkvh, H_, QKV_N);
  // 4. sliding-window flash attention
  attn_kernel<<<dim3(S_ / 16, NH_), 32, 0, stream>>>(qkvh, oh);
  // 5. x2 = o @ w_o^T + residual -> residual2 output
  gemm_oproj_kernel<<<dim3(S_ / BM, H_ / BN), 256, 0, stream>>>(
      oh, w_o, hidden, res2, NH_ * HD_);
  // 6. rmsnorm2 -> f16
  rmsnorm_cast_kernel<<<dim3(S_), 256, 0, stream>>>(res2, ln2_w, th);
  // 7. router + top-2 token lists
  router_kernel<<<dim3(S_ / 256), 256, 0, stream>>>(th, router_w, counts, list, gates);
  // 8. exclusive scan of expert counts
  scan_kernel<<<dim3(1), 32, 0, stream>>>(counts, offsets);
  // 9. gated up-proj + silu (sparse, gathered, async-LDS staging)
  gemm_gated_kernel<<<dim3(S_ / BM, I_ / BN, E_), 256, 0, stream>>>(
      th, ws, counts, offsets, list, abuf);
  // 10. down-proj, gate-scaled atomic scatter into out
  gemm_y_kernel<<<dim3(S_ / BM, H_ / BN, E_), 256, 0, stream>>>(
      abuf, w2s, counts, offsets, list, gates, out);
}
